// MultiHeadAttention_64450279243850
// MI455X (gfx1250) — matrix operand order repaired and checked by reading, not compiled, not run on hardware
//
#include <hip/hip_runtime.h>

// ---------------- CDNA5 WMMA helpers ----------------
typedef __bf16 bf16x16 __attribute__((ext_vector_type(16)));
typedef float  f32x8   __attribute__((ext_vector_type(8)));

union FragAB { bf16x16 v; unsigned int u[8]; };
union FragC  { f32x8 v; float f[8]; };

__device__ __forceinline__ unsigned short f2bf(float x) {
  unsigned int u = __float_as_uint(x);
  unsigned int r = u + 0x7FFFu + ((u >> 16) & 1u);   // round-to-nearest-even
  return (unsigned short)(r >> 16);
}

// A-matrix (16x32, 16-bit) K index for VGPR v, lane-half hf (two consecutive K per VGPR)
__device__ __forceinline__ int kA(int v, int hf) {
  return ((v & 3) * 2) + hf * 8 + ((v >> 2) * 16);
}
// B-matrix (32x16, 16-bit) K index for VGPR v, lane-half hf
__device__ __forceinline__ int kB(int v, int hf) {
  return ((v & 3) * 2) + hf * 8 + ((v >> 2) * 16);
}

// CDNA5 async global->LDS copy (16B per lane), tracked by ASYNCcnt.
// LDS byte address = low 32 bits of generic shared pointer (flat aperture mapping).
#define ASYNC_LOAD_B128(ldsPtr, gPtr)                                          \
  asm volatile("global_load_async_to_lds_b128 %0, %1, off"                     \
               :: "v"((unsigned)(unsigned long long)(ldsPtr)),                 \
                  "v"((unsigned long long)(gPtr))                              \
               : "memory")
#define WAIT_ASYNCCNT0() asm volatile("s_wait_asynccnt 0x0" ::: "memory")

// Problem constants: B=4, S=2048, D_MODEL=1024, H=16, D_HEAD=64, BS=8192

// ---------------- Kernel 1: QKV projections ----------------
// grid = (64, 16, 3): 128-row x 64-col tiles over [8192,1024]; z selects Q/K/V.
// Q/K -> bf16 head layout [B,H,S,64] (Q pre-scaled by 1/sqrt(64));
// V   -> bf16 transposed  [B,H,64,S] so attention can async-stage it directly.
__global__ __launch_bounds__(256) void qkv_gemm(
    const float* __restrict__ x,
    const float* __restrict__ Wq, const float* __restrict__ bq,
    const float* __restrict__ Wk, const float* __restrict__ bk,
    const float* __restrict__ Wv, const float* __restrict__ bv,
    unsigned short* __restrict__ Qb, unsigned short* __restrict__ Kb,
    unsigned short* __restrict__ Vt)
{
  __shared__ __align__(16) unsigned short Asm[128 * 32];
  __shared__ __align__(16) unsigned short Bsm[64 * 32];

  const int t = threadIdx.x;
  const int w = t >> 5, L = t & 31, lid = L & 15, hf = L >> 4;
  const int row0 = blockIdx.x * 128, col0 = blockIdx.y * 64;
  const int z = blockIdx.z;
  const float* W    = (z == 0) ? Wq : (z == 1) ? Wk : Wv;
  const float* bias = (z == 0) ? bq : (z == 1) ? bk : bv;
  unsigned short* outp = (z == 0) ? Qb : (z == 1) ? Kb : Vt;
  const float scale = (z == 0) ? 0.125f : 1.0f;

  FragC acc[4];
  #pragma unroll
  for (int n = 0; n < 4; n++)
    #pragma unroll
    for (int i = 0; i < 8; i++) acc[n].f[i] = 0.f;

  for (int kk = 0; kk < 1024; kk += 32) {
    __syncthreads();
    // stage A: 128x32 f32 -> bf16, row-major [r][k]
    #pragma unroll
    for (int i = 0; i < 16; i++) {
      int idx = t + i * 256;
      int r = idx >> 5, c = idx & 31;
      Asm[r * 32 + c] = f2bf(x[(size_t)(row0 + r) * 1024 + kk + c]);
    }
    // stage B: 32x64 f32 -> bf16 stored [n][k] so K pairs are contiguous
    #pragma unroll
    for (int i = 0; i < 8; i++) {
      int idx = t + i * 256;
      int k = idx >> 6, n = idx & 63;
      Bsm[n * 32 + k] = f2bf(W[(size_t)(kk + k) * 1024 + col0 + n]);
    }
    __syncthreads();

    FragAB a;
    #pragma unroll
    for (int v = 0; v < 8; v++)
      a.u[v] = *(const unsigned int*)&Asm[(w * 16 + lid) * 32 + kA(v, hf)];
    #pragma unroll
    for (int nt = 0; nt < 4; nt++) {
      FragAB bfr;
      #pragma unroll
      for (int v = 0; v < 8; v++)
        bfr.u[v] = *(const unsigned int*)&Bsm[(nt * 16 + lid) * 32 + kB(v, hf)];
      acc[nt].v = __builtin_amdgcn_wmma_f32_16x16x32_bf16(
          false, a.v, false, bfr.v, (short)0, acc[nt].v, false, false);
    }
  }

  // epilogue
  #pragma unroll
  for (int nt = 0; nt < 4; nt++) {
    #pragma unroll
    for (int v = 0; v < 8; v++) {
      int r = row0 + w * 16 + v + 8 * hf;
      int c = col0 + nt * 16 + lid;
      float val = (acc[nt].f[v] + bias[c]) * scale;
      int h = c >> 6, d = c & 63;
      int bi = r >> 11, s = r & 2047;
      size_t bh = (size_t)bi * 16 + h;
      if (z == 2)   // V transposed: [B,H,64,S]
        outp[(bh * 64 + d) * 2048 + s] = f2bf(val);
      else          // Q/K head layout: [B,H,S,64]
        outp[(bh * 2048 + s) * 64 + d] = f2bf(val);
    }
  }
}

// ---------------- Kernel 2: flash attention ----------------
// grid = (16, 64): blockIdx.y = b*H + h, each block does 128 query rows (16/wave).
__global__ __launch_bounds__(256) void attn_kernel(
    const unsigned short* __restrict__ Qb,
    const unsigned short* __restrict__ Kb,
    const unsigned short* __restrict__ Vt,
    const unsigned char* __restrict__ mask,
    unsigned short* __restrict__ Ctx)
{
  __shared__ __align__(16) unsigned short Ks[32 * 64];       // [key][d]
  __shared__ __align__(16) unsigned short VsT[64 * 32];      // [d][key]
  __shared__ __align__(16) unsigned short Ps[8 * 16 * 32];   // per-wave P tiles

  const int t = threadIdx.x;
  const int w = t >> 5, L = t & 31, lid = L & 15, hf = L >> 4;
  const int bh = blockIdx.y;
  const int b = bh >> 4, h = bh & 15;
  const int q0 = blockIdx.x * 128;
  const size_t kvbase = (size_t)bh * 2048 * 64;

  // Q fragments (two K=32 chunks over d_head=64), loaded once
  FragAB qf[2];
  {
    int srow = q0 + w * 16 + lid;
    const unsigned short* qp = Qb + kvbase + (size_t)srow * 64;
    #pragma unroll
    for (int cc = 0; cc < 2; cc++)
      #pragma unroll
      for (int v = 0; v < 8; v++)
        qf[cc].u[v] = *(const unsigned int*)&qp[cc * 32 + kA(v, hf)];
  }

  FragC O[4];
  float mrow[8], lrow[8];
  #pragma unroll
  for (int i = 0; i < 4; i++)
    #pragma unroll
    for (int j = 0; j < 8; j++) O[i].f[j] = 0.f;
  #pragma unroll
  for (int v = 0; v < 8; v++) { mrow[v] = -__builtin_inff(); lrow[v] = 0.f; }

  // per-thread async staging addresses (16B segments, 256 segs per tile)
  //   K tile: 32 keys x 64 d contiguous  -> seg t copies elements [t*8, t*8+8)
  //   V tile: VsT[d][key], d = t>>2, key chunk = (t&3)*8, source row stride S
  const int vd = t >> 2, vk = (t & 3) * 8;

  for (int kb = 0; kb < 64; kb++) {
    __syncthreads();   // previous iteration's LDS reads complete
    ASYNC_LOAD_B128(&Ks[t * 8], Kb + kvbase + (size_t)kb * 32 * 64 + t * 8);
    ASYNC_LOAD_B128(&VsT[vd * 32 + vk],
                    Vt + kvbase + (size_t)vd * 2048 + kb * 32 + vk);
    WAIT_ASYNCCNT0();
    __syncthreads();

    // scores: two 16-key subtiles, accumulate over d in two K=32 WMMAs each
    FragC sA, sB;
    #pragma unroll
    for (int i = 0; i < 8; i++) { sA.f[i] = 0.f; sB.f[i] = 0.f; }
    #pragma unroll
    for (int cc = 0; cc < 2; cc++) {
      FragAB kf;
      #pragma unroll
      for (int v = 0; v < 8; v++)
        kf.u[v] = *(const unsigned int*)&Ks[lid * 64 + cc * 32 + kB(v, hf)];
      sA.v = __builtin_amdgcn_wmma_f32_16x16x32_bf16(
          false, qf[cc].v, false, kf.v, (short)0, sA.v, false, false);
      #pragma unroll
      for (int v = 0; v < 8; v++)
        kf.u[v] = *(const unsigned int*)&Ks[(16 + lid) * 64 + cc * 32 + kB(v, hf)];
      sB.v = __builtin_amdgcn_wmma_f32_16x16x32_bf16(
          false, qf[cc].v, false, kf.v, (short)0, sB.v, false, false);
    }

    // key mask (this lane's two key columns)
    int jg0 = kb * 32 + lid;
    bool m0 = mask[(size_t)b * 2048 + jg0] != 0;
    bool m1 = mask[(size_t)b * 2048 + jg0 + 16] != 0;
    #pragma unroll
    for (int v = 0; v < 8; v++) {
      if (!m0) sA.f[v] = -__builtin_inff();
      if (!m1) sB.f[v] = -__builtin_inff();
    }

    // online softmax per row (rows live across the 16-lane half-groups)
    #pragma unroll
    for (int v = 0; v < 8; v++) {
      float a0 = sA.f[v], a1 = sB.f[v];
      float rmax = fmaxf(a0, a1);
      #pragma unroll
      for (int off = 1; off < 16; off <<= 1)
        rmax = fmaxf(rmax, __shfl_xor(rmax, off, 32));
      float mprev = mrow[v];
      float mnew = fmaxf(mprev, rmax);
      bool ninf = (mnew == -__builtin_inff());
      float corr = ninf ? 1.f : __expf(mprev - mnew);
      float p0 = ninf ? 0.f : __expf(a0 - mnew);
      float p1 = ninf ? 0.f : __expf(a1 - mnew);
      float rsum = p0 + p1;
      #pragma unroll
      for (int off = 1; off < 16; off <<= 1)
        rsum += __shfl_xor(rsum, off, 32);
      lrow[v] = lrow[v] * corr + rsum;
      mrow[v] = mnew;
      #pragma unroll
      for (int tt = 0; tt < 4; tt++) O[tt].f[v] *= corr;
      int prow = v + 8 * hf;                 // C-layout row -> P tile row
      Ps[(w * 16 + prow) * 32 + lid]      = f2bf(p0);
      Ps[(w * 16 + prow) * 32 + lid + 16] = f2bf(p1);
    }
    __syncthreads();   // C-layout -> A-layout re-shuffle via LDS

    // P x V: one 16x32 A fragment, four d-subtiles of V
    FragAB pf;
    #pragma unroll
    for (int v = 0; v < 8; v++)
      pf.u[v] = *(const unsigned int*)&Ps[(w * 16 + lid) * 32 + kA(v, hf)];
    #pragma unroll
    for (int tt = 0; tt < 4; tt++) {
      FragAB vf;
      #pragma unroll
      for (int v = 0; v < 8; v++)
        vf.u[v] = *(const unsigned int*)&VsT[(tt * 16 + lid) * 32 + kB(v, hf)];
      O[tt].v = __builtin_amdgcn_wmma_f32_16x16x32_bf16(
          false, pf.v, false, vf.v, (short)0, O[tt].v, false, false);
    }
  }

  // epilogue: normalize, write context bf16 [B,S,1024]
  #pragma unroll
  for (int tt = 0; tt < 4; tt++) {
    #pragma unroll
    for (int v = 0; v < 8; v++) {
      float inv = (lrow[v] > 0.f) ? 1.f / lrow[v] : 0.f;
      int srow = q0 + w * 16 + v + 8 * hf;
      int c = h * 64 + tt * 16 + lid;
      Ctx[((size_t)b * 2048 + srow) * 1024 + c] = f2bf(O[tt].f[v] * inv);
    }
  }
}

// ---------------- Kernel 3: output projection ----------------
// grid = (64, 16): Ctx bf16 [8192,1024] x Wo -> f32 out + bo
__global__ __launch_bounds__(256) void out_gemm(
    const unsigned short* __restrict__ Actx,
    const float* __restrict__ Wo, const float* __restrict__ bo,
    float* __restrict__ out)
{
  __shared__ __align__(16) unsigned short Asm[128 * 32];
  __shared__ __align__(16) unsigned short Bsm[64 * 32];

  const int t = threadIdx.x;
  const int w = t >> 5, L = t & 31, lid = L & 15, hf = L >> 4;
  const int row0 = blockIdx.x * 128, col0 = blockIdx.y * 64;

  FragC acc[4];
  #pragma unroll
  for (int n = 0; n < 4; n++)
    #pragma unroll
    for (int i = 0; i < 8; i++) acc[n].f[i] = 0.f;

  for (int kk = 0; kk < 1024; kk += 32) {
    __syncthreads();
    // stage A (already bf16): async 16B copies, 128x32 = 512 segments, 2 per thread
    {
      int r = t >> 1, c8 = (t & 1) * 8;   // 16 bf16 per row = 2 segments
      ASYNC_LOAD_B128(&Asm[r * 32 + c8],
                      Actx + (size_t)(row0 + r) * 1024 + kk + c8);
      int r2 = r, c8b = c8 + 16;
      ASYNC_LOAD_B128(&Asm[r2 * 32 + c8b],
                      Actx + (size_t)(row0 + r2) * 1024 + kk + c8b);
    }
    // stage B: Wo f32 -> bf16 [n][k]
    #pragma unroll
    for (int i = 0; i < 8; i++) {
      int idx = t + i * 256;
      int k = idx >> 6, n = idx & 63;
      Bsm[n * 32 + k] = f2bf(Wo[(size_t)(kk + k) * 1024 + col0 + n]);
    }
    WAIT_ASYNCCNT0();
    __syncthreads();

    FragAB a;
    #pragma unroll
    for (int v = 0; v < 8; v++)
      a.u[v] = *(const unsigned int*)&Asm[(w * 16 + lid) * 32 + kA(v, hf)];
    #pragma unroll
    for (int nt = 0; nt < 4; nt++) {
      FragAB bfr;
      #pragma unroll
      for (int v = 0; v < 8; v++)
        bfr.u[v] = *(const unsigned int*)&Bsm[(nt * 16 + lid) * 32 + kB(v, hf)];
      acc[nt].v = __builtin_amdgcn_wmma_f32_16x16x32_bf16(
          false, a.v, false, bfr.v, (short)0, acc[nt].v, false, false);
    }
  }

  #pragma unroll
  for (int nt = 0; nt < 4; nt++)
    #pragma unroll
    for (int v = 0; v < 8; v++) {
      int r = row0 + w * 16 + v + 8 * hf;
      int c = col0 + nt * 16 + lid;
      out[(size_t)r * 1024 + c] = acc[nt].f[v] + bo[c];
    }
}

// ---------------- host launcher ----------------
extern "C" void kernel_launch(void* const* d_in, const int* in_sizes, int n_in,
                              void* d_out, int out_size, void* d_ws, size_t ws_size,
                              hipStream_t stream) {
  const float* x          = (const float*)d_in[0];
  const unsigned char* mk = (const unsigned char*)d_in[1];
  const float* Wq = (const float*)d_in[2];
  const float* bq = (const float*)d_in[3];
  const float* Wk = (const float*)d_in[4];
  const float* bk = (const float*)d_in[5];
  const float* Wv = (const float*)d_in[6];
  const float* bv = (const float*)d_in[7];
  const float* Wo = (const float*)d_in[8];
  const float* bo = (const float*)d_in[9];
  (void)in_sizes; (void)n_in; (void)out_size; (void)ws_size;

  // workspace: Q,K [B,H,S,64], V transposed [B,H,64,S], Ctx [B,S,1024] (all bf16) = 64 MB
  const size_t nElem = (size_t)4 * 16 * 2048 * 64;
  unsigned short* Qb  = (unsigned short*)d_ws;
  unsigned short* Kb  = Qb + nElem;
  unsigned short* Vt  = Kb + nElem;
  unsigned short* Ctx = Vt + nElem;

  qkv_gemm<<<dim3(64, 16, 3), 256, 0, stream>>>(x, Wq, bq, Wk, bk, Wv, bv, Qb, Kb, Vt);
  attn_kernel<<<dim3(16, 64), 256, 0, stream>>>(Qb, Kb, Vt, mk, Ctx);
  out_gemm<<<dim3(64, 16), 256, 0, stream>>>(Ctx, Wo, bo, (float*)d_out);
}
